// PhasedLSTMModel_48430051230323
// MI455X (gfx1250) — matrix operand order repaired and checked by reading, not compiled, not run on hardware
//
#include <hip/hip_runtime.h>
#include <hip/hip_bf16.h>

typedef _Float16 f16;
typedef __attribute__((ext_vector_type(16))) _Float16 v16h;
typedef __attribute__((ext_vector_type(8)))  _Float16 v8h;
typedef __attribute__((ext_vector_type(8)))  float    v8f;

#define BB   256
#define TT   256
#define II   64
#define HH   512
#define DD   256
#define R_ON_C   0.05f
#define ALPHA_C  1e-3f
#define LN_EPS_C 1e-5f

__device__ __forceinline__ float sigm(float x)     { return 1.0f / (1.0f + __expf(-x)); }
__device__ __forceinline__ float tanhfast(float x) { return 1.0f - 2.0f / (__expf(2.0f * x) + 1.0f); }

__device__ __forceinline__ v8f splat8(float v) {
  v8f r;
#pragma unroll
  for (int i = 0; i < 8; ++i) r[i] = v;
  return r;
}

// Load a 16x32 f16 A-tile (M x K) from a row-major [16][stride] f16 buffer.
// A layout (matching B): lanes 0-15 row M=lane, K 0-15;
// lanes 16-31 hold K 16-31. -> two contiguous 16B LDS loads per lane.
__device__ __forceinline__ v16h load_a16(const f16* base, int stride, int kc, int lane) {
  const int m    = lane & 15;
  const int koff = (lane >> 4) << 4;          // 0 or 16
  const f16* p = base + m * stride + kc + koff;
  v8h lo = *(const v8h*)p;                    // K = kc+koff .. +7
  v8h hi = *(const v8h*)(p + 8);              // K = kc+koff+8 .. +15
  v16h a;
#pragma unroll
  for (int i = 0; i < 8; ++i) { a[i] = lo[i]; a[i + 8] = hi[i]; }
  return a;
}

__device__ __forceinline__ v8f wmma16(v16h a, v16h b, v8f c) {
  return __builtin_amdgcn_wmma_f32_16x16x32_f16(false, a, false, b, (short)0, c, false, false);
}

// ---------------- prep kernels ----------------

// U [H][4H] f32 -> Ut [4H][H] f16 (row-major). Ut row n, col k = U[k][n].
__global__ __launch_bounds__(256) void pack_u_kernel(const float* __restrict__ U, f16* __restrict__ Ut) {
  int idx = blockIdx.x * 256 + threadIdx.x;
  if (idx < 4 * HH * HH) {
    int n = idx / HH, k = idx - n * HH;
    Ut[idx] = (f16)U[(size_t)k * (4 * HH) + n];
  }
}

// W [I][4H] f32 -> Wt [4H][I] f16.
__global__ __launch_bounds__(256) void pack_w_kernel(const float* __restrict__ W, f16* __restrict__ Wt) {
  int idx = blockIdx.x * 256 + threadIdx.x;
  if (idx < 4 * HH * II) {
    int n = idx / II, k = idx - n * II;
    Wt[idx] = (f16)W[(size_t)k * (4 * HH) + n];
  }
}

// Fold dense head: v[j] = sum_d W1[j][d]*W2[d];  cs = b1.W2 + b2
__global__ __launch_bounds__(256) void fold_head_kernel(const float* __restrict__ W1, const float* __restrict__ b1,
                                                        const float* __restrict__ W2, const float* __restrict__ b2,
                                                        float* __restrict__ v, float* __restrict__ cs) {
  int j = blockIdx.x * 256 + threadIdx.x;
  if (j < HH) {
    float acc = 0.f;
    for (int d = 0; d < DD; ++d) acc += W1[(size_t)j * DD + d] * W2[d];
    v[j] = acc;
  }
  if (j == 0) {
    float acc = 0.f;
    for (int d = 0; d < DD; ++d) acc += b1[d] * W2[d];
    cs[0] = acc + b2[0];
  }
}

// ---------------- main persistent phased-LSTM kernel ----------------
// grid = B/16 workgroups, block = 512 threads (16 waves).
// Wave w owns hidden tiles [w*32, w*32+16) and [w*32+16, w*32+32).
__global__ __launch_bounds__(512)
void plstm_persistent_kernel(const float* __restrict__ x,      // [B][T][I] f32
                             const f16*  __restrict__ Ut,      // [4H][H] f16
                             const f16*  __restrict__ Wt,      // [4H][I] f16
                             const float* __restrict__ bias,   // [4H]
                             const float* __restrict__ tauv,   // [H]
                             const float* __restrict__ sph,    // [H]
                             float* __restrict__ h_out)        // [B][H] f32
{
  __shared__ __align__(16) f16 lds_h[16 * HH];   // 16 KB: h_t for this batch tile
  __shared__ __align__(16) f16 lds_x[16 * II];   // 2 KB: x_t for this batch tile

  const int tid   = threadIdx.x;
  const int wave  = tid >> 5;
  const int lane  = tid & 31;
  const int lhalf = lane >> 4;                 // 0/1: which half of the wave
  const int b0    = blockIdx.x << 4;           // batch-row base
  const int j0    = (wave << 5) + (lane & 15); // hidden unit, tile 0
  const int j1    = j0 + 16;                   // hidden unit, tile 1

  // per-lane constants (N column is fixed per lane in the C layout)
  const float bi0 = bias[0 * HH + j0], bf0 = bias[1 * HH + j0], bg0 = bias[2 * HH + j0], bo0 = bias[3 * HH + j0];
  const float bi1 = bias[0 * HH + j1], bf1 = bias[1 * HH + j1], bg1 = bias[2 * HH + j1], bo1 = bias[3 * HH + j1];
  const float tau0 = tauv[j0], s0 = sph[j0];
  const float tau1 = tauv[j1], s1 = sph[j1];

  // B-tile base pointers: this code's B order (same as load_a16) gives each lane 16 contiguous K values
  // of its column; with Ut/Wt stored transposed row-major that is one 32B load.
  const f16* ub0 = Ut + (size_t)j0 * HH + (lhalf << 4);
  const f16* ub1 = Ut + (size_t)j1 * HH + (lhalf << 4);
  const f16* wb0 = Wt + (size_t)j0 * II + (lhalf << 4);
  const f16* wb1 = Wt + (size_t)j1 * II + (lhalf << 4);

  for (int i = tid; i < 16 * HH; i += 512) lds_h[i] = (f16)0.0f;

  v8f c0 = splat8(0.f), c1 = splat8(0.f);
  v8f h0 = splat8(0.f), h1 = splat8(0.f);

  for (int t = 0; t < TT; ++t) {
    // Opaque zero: volatile asm is never hoisted by LICM, so every weight
    // pointer below is loop-variant -> B-tile loads stay inside the t-loop
    // (they hit L2 every step) instead of being hoisted and spilled to scratch.
    int tz;
    asm volatile("s_mov_b32 %0, 0" : "=s"(tz));
    const f16* ub0t = ub0 + tz;
    const f16* ub1t = ub1 + tz;
    const f16* wb0t = wb0 + tz;
    const f16* wb1t = wb1 + tz;

    // stage x_t (16 rows x 64 cols) into LDS as f16
    for (int i = tid; i < 16 * II; i += 512) {
      int m = i >> 6, col = i & 63;
      lds_x[i] = (f16)x[((size_t)(b0 + m) * TT + t) * II + col];
    }
    __syncthreads();   // x_t staged AND previous-step h stores visible

    v8f zi0 = splat8(bi0), zf0 = splat8(bf0), zg0 = splat8(bg0), zo0 = splat8(bo0);
    v8f zi1 = splat8(bi1), zf1 = splat8(bf1), zg1 = splat8(bg1), zo1 = splat8(bo1);

    // z += x_t @ W   (K = 64)
#pragma unroll
    for (int kc = 0; kc < II; kc += 32) {
      v16h a = load_a16(lds_x, II, kc, lane);
      zi0 = wmma16(a, *(const v16h*)(wb0t + (size_t)0 * HH * II + kc), zi0);
      zf0 = wmma16(a, *(const v16h*)(wb0t + (size_t)1 * HH * II + kc), zf0);
      zg0 = wmma16(a, *(const v16h*)(wb0t + (size_t)2 * HH * II + kc), zg0);
      zo0 = wmma16(a, *(const v16h*)(wb0t + (size_t)3 * HH * II + kc), zo0);
      zi1 = wmma16(a, *(const v16h*)(wb1t + (size_t)0 * HH * II + kc), zi1);
      zf1 = wmma16(a, *(const v16h*)(wb1t + (size_t)1 * HH * II + kc), zf1);
      zg1 = wmma16(a, *(const v16h*)(wb1t + (size_t)2 * HH * II + kc), zg1);
      zo1 = wmma16(a, *(const v16h*)(wb1t + (size_t)3 * HH * II + kc), zo1);
    }
    // z += h @ U     (K = 512), A chunk shared by all 8 WMMAs
#pragma unroll 2
    for (int kc = 0; kc < HH; kc += 32) {
      v16h a = load_a16(lds_h, HH, kc, lane);
      zi0 = wmma16(a, *(const v16h*)(ub0t + (size_t)0 * HH * HH + kc), zi0);
      zf0 = wmma16(a, *(const v16h*)(ub0t + (size_t)1 * HH * HH + kc), zf0);
      zg0 = wmma16(a, *(const v16h*)(ub0t + (size_t)2 * HH * HH + kc), zg0);
      zo0 = wmma16(a, *(const v16h*)(ub0t + (size_t)3 * HH * HH + kc), zo0);
      zi1 = wmma16(a, *(const v16h*)(ub1t + (size_t)0 * HH * HH + kc), zi1);
      zf1 = wmma16(a, *(const v16h*)(ub1t + (size_t)1 * HH * HH + kc), zf1);
      zg1 = wmma16(a, *(const v16h*)(ub1t + (size_t)2 * HH * HH + kc), zg1);
      zo1 = wmma16(a, *(const v16h*)(ub1t + (size_t)3 * HH * HH + kc), zo1);
    }

    // time gate k(t) is per hidden unit -> per-lane scalar per tile
    const float tf = (float)t;
    float m0 = fmodf(tf - s0, tau0); m0 += (m0 < 0.f) ? tau0 : 0.f;
    float m1 = fmodf(tf - s1, tau1); m1 += (m1 < 0.f) ? tau1 : 0.f;
    const float phi0 = m0 / tau0, phi1 = m1 / tau1;
    const float k0 = (phi0 < 0.5f * R_ON_C) ? (2.f * phi0 / R_ON_C)
                   : (phi0 < R_ON_C)        ? (2.f - 2.f * phi0 / R_ON_C)
                                            : (ALPHA_C * phi0);
    const float k1 = (phi1 < 0.5f * R_ON_C) ? (2.f * phi1 / R_ON_C)
                   : (phi1 < R_ON_C)        ? (2.f - 2.f * phi1 / R_ON_C)
                                            : (ALPHA_C * phi1);

#pragma unroll
    for (int r = 0; r < 8; ++r) {
      { // tile 0
        float ig = sigm(zi0[r]), fg = sigm(zf0[r]);
        float gg = tanhfast(zg0[r]), og = sigm(zo0[r]);
        float ct = fg * c0[r] + ig * gg;
        float ht = og * tanhfast(ct);
        c0[r] = k0 * ct + (1.f - k0) * c0[r];
        h0[r] = k0 * ht + (1.f - k0) * h0[r];
      }
      { // tile 1
        float ig = sigm(zi1[r]), fg = sigm(zf1[r]);
        float gg = tanhfast(zg1[r]), og = sigm(zo1[r]);
        float ct = fg * c1[r] + ig * gg;
        float ht = og * tanhfast(ct);
        c1[r] = k1 * ct + (1.f - k1) * c1[r];
        h1[r] = k1 * ht + (1.f - k1) * h1[r];
      }
    }

    __syncthreads();   // all waves done reading old lds_h
#pragma unroll
    for (int r = 0; r < 8; ++r) {
      int mrow = r + (lhalf << 3);           // C layout: VGPR r -> row r or r+8
      lds_h[mrow * HH + j0] = (f16)h0[r];
      lds_h[mrow * HH + j1] = (f16)h1[r];
    }
  }

  // write last h (f32, from registers) for the head
#pragma unroll
  for (int r = 0; r < 8; ++r) {
    int mrow = r + (lhalf << 3);
    h_out[(size_t)(b0 + mrow) * HH + j0] = h0[r];
    h_out[(size_t)(b0 + mrow) * HH + j1] = h1[r];
  }
}

// ---------------- head: LayerNorm + folded dot ----------------
__global__ __launch_bounds__(256)
void head_ln_dot_kernel(const float* __restrict__ h, const float* __restrict__ gamma,
                        const float* __restrict__ beta, const float* __restrict__ v,
                        const float* __restrict__ cs, float* __restrict__ out) {
  __shared__ float red[256];
  const int row = blockIdx.x, tid = threadIdx.x;
  const float* hr = h + (size_t)row * HH;
  const float a = hr[tid], b = hr[tid + 256];

  red[tid] = a + b;
  __syncthreads();
  for (int off = 128; off > 0; off >>= 1) { if (tid < off) red[tid] += red[tid + off]; __syncthreads(); }
  const float mu = red[0] * (1.0f / HH);
  __syncthreads();

  const float d0 = a - mu, d1 = b - mu;
  red[tid] = d0 * d0 + d1 * d1;
  __syncthreads();
  for (int off = 128; off > 0; off >>= 1) { if (tid < off) red[tid] += red[tid + off]; __syncthreads(); }
  const float inv = rsqrtf(red[0] * (1.0f / HH) + LN_EPS_C);
  __syncthreads();

  const float hn0 = d0 * inv * gamma[tid] + beta[tid];
  const float hn1 = d1 * inv * gamma[tid + 256] + beta[tid + 256];
  red[tid] = hn0 * v[tid] + hn1 * v[tid + 256];
  __syncthreads();
  for (int off = 128; off > 0; off >>= 1) { if (tid < off) red[tid] += red[tid + off]; __syncthreads(); }
  if (tid == 0) out[row] = red[0] + cs[0];
}

// ---------------- launcher ----------------
extern "C" void kernel_launch(void* const* d_in, const int* in_sizes, int n_in,
                              void* d_out, int out_size, void* d_ws, size_t ws_size,
                              hipStream_t stream) {
  (void)in_sizes; (void)n_in; (void)out_size; (void)ws_size;
  const float* x    = (const float*)d_in[0];
  const float* W    = (const float*)d_in[1];
  const float* U    = (const float*)d_in[2];
  const float* b    = (const float*)d_in[3];
  const float* tau  = (const float*)d_in[4];
  const float* s    = (const float*)d_in[5];
  const float* ln_g = (const float*)d_in[6];
  const float* ln_b = (const float*)d_in[7];
  const float* W1   = (const float*)d_in[8];
  const float* b1   = (const float*)d_in[9];
  const float* W2   = (const float*)d_in[10];
  const float* b2   = (const float*)d_in[11];

  char* ws = (char*)d_ws;
  size_t off = 0;
  f16* Ut = (f16*)(ws + off);            off += (size_t)4 * HH * HH * sizeof(f16);  // 2 MB
  f16* Wt = (f16*)(ws + off);            off += (size_t)4 * HH * II * sizeof(f16);  // 256 KB
  float* vfold = (float*)(ws + off);     off += (size_t)HH * sizeof(float);
  float* csum  = (float*)(ws + off);     off += 16;                                 // keep 16B align
  float* hlast = (float*)(ws + off);     off += (size_t)BB * HH * sizeof(float);    // 512 KB

  pack_u_kernel<<<(4 * HH * HH + 255) / 256, 256, 0, stream>>>(U, Ut);
  pack_w_kernel<<<(4 * HH * II + 255) / 256, 256, 0, stream>>>(W, Wt);
  fold_head_kernel<<<2, 256, 0, stream>>>(W1, b1, W2, b2, vfold, csum);

  plstm_persistent_kernel<<<BB / 16, 512, 0, stream>>>(x, Ut, Wt, b, tau, s, hlast);

  head_ln_dot_kernel<<<BB, 256, 0, stream>>>(hlast, ln_g, ln_b, vfold, csum, (float*)d_out);
}
